// LanguageModel_57234734187002
// MI455X (gfx1250) — compile-verified, matrix operand order checked by reading, not run on hardware
//
#include <hip/hip_runtime.h>
#include <math.h>

// ---------------- problem constants ----------------
constexpr int V  = 32000;
constexpr int E  = 400;
constexpr int H  = 1150;
constexpr int B  = 16;
constexpr int T  = 128;
constexpr int M  = B * T;          // 2048 time-batched rows (row m = t*16 + b)

// padded dims: K padded to multiple of 32 (WMMA K), N padded to multiple of 64
constexpr int Ep  = 416;           // E  -> 416  (13 * 32)
constexpr int Hp  = 1152;          // H  -> 1152 (36 * 32)
constexpr int N1  = 4 * H;         // 4600 gate columns (layers 1,2)
constexpr int N1p = 4608;          // 72 * 64
constexpr int N3  = 4 * E;         // 1600 gate columns (layer 3), 25 * 64
constexpr int LDSW = 1152;         // fixed LDS row stride for recurrent h

typedef __bf16 bf16_t;
typedef __attribute__((ext_vector_type(16))) __bf16 v16bf;
typedef __attribute__((ext_vector_type(8)))  __bf16 v8bf;
typedef __attribute__((ext_vector_type(8)))  float  v8f;

// ---------------- fragment load (CDNA5 ISA 7.12.2, 16-bit A/B 16x32) ----------------
// lanes 0-15: K 0..7 and 16..23; lanes 16-31: K 8..15 and 24..31
// Works for both global and LDS (generic) pointers: emits global_load_b128 / ds_load_b128.
__device__ inline v16bf load_frag(const bf16_t* row, int k0, int half_id) {
    v8bf lo = *(const v8bf*)(row + k0 + half_id * 8);
    v8bf hi = *(const v8bf*)(row + k0 + 16 + half_id * 8);
    v16bf f;
#pragma unroll
    for (int i = 0; i < 8; ++i) { f[i] = lo[i]; f[i + 8] = hi[i]; }
    return f;
}

__device__ inline float sigmoidf_fast(float x) { return 1.f / (1.f + __expf(-x)); }

// ---------------- register-blocked GEMM: C = A * Bw^T (+bias) ----------------
// Wave macro-tile: (16*MT) x (16*NT). 4 waves/block tile N.
// A: row-major (M x K) bf16. Bw: row-major (N x K) bf16 (i.e. B^T). SCATTER writes (B,V,T).
template <int MT, int NT, int SCATTER>
__global__ __launch_bounds__(128) void gemm_bf16_wmma(
    const bf16_t* __restrict__ A, int lda,
    const bf16_t* __restrict__ Bw, int ldb,
    const float* __restrict__ bias0, const float* __restrict__ bias1, int nbias,
    float* __restrict__ C, int ldc, int K, int Np) {
    const int lane    = threadIdx.x & 31;
    const int wave    = threadIdx.x >> 5;
    const int lr      = lane & 15;
    const int half_id = lane >> 4;
    const int n0 = (blockIdx.x * 4 + wave) * (16 * NT);
    const int m0 = blockIdx.y * (16 * MT);
    if (n0 >= Np) return;  // partial-coverage wave; no barriers used, safe to exit

    const bf16_t* arow[MT];
    const bf16_t* brow[NT];
#pragma unroll
    for (int mt = 0; mt < MT; ++mt) arow[mt] = A + (size_t)(m0 + mt * 16 + lr) * lda;
#pragma unroll
    for (int nt = 0; nt < NT; ++nt) brow[nt] = Bw + (size_t)(n0 + nt * 16 + lr) * ldb;

    v8f acc[MT][NT];
#pragma unroll
    for (int mt = 0; mt < MT; ++mt)
#pragma unroll
        for (int nt = 0; nt < NT; ++nt) acc[mt][nt] = v8f{};

    for (int k0 = 0; k0 < K; k0 += 32) {
        __builtin_prefetch(brow[0] + k0 + 64, 0, 1);   // weight stream, 2 chunks ahead
        v16bf a[MT], b[NT];
#pragma unroll
        for (int mt = 0; mt < MT; ++mt) a[mt] = load_frag(arow[mt], k0, half_id);
#pragma unroll
        for (int nt = 0; nt < NT; ++nt) b[nt] = load_frag(brow[nt], k0, half_id);
#pragma unroll
        for (int nt = 0; nt < NT; ++nt)
#pragma unroll
            for (int mt = 0; mt < MT; ++mt)
                acc[mt][nt] = __builtin_amdgcn_wmma_f32_16x16x32_bf16(
                    false, a[mt], false, b[nt], (short)0, acc[mt][nt], false, false);
    }

#pragma unroll
    for (int mt = 0; mt < MT; ++mt) {
#pragma unroll
        for (int nt = 0; nt < NT; ++nt) {
            const int n = n0 + nt * 16 + lr;
            float bias = 0.f;
            if (n < nbias) {
                if (bias0) bias += bias0[n];
                if (bias1) bias += bias1[n];
            }
#pragma unroll
            for (int r = 0; r < 8; ++r) {
                // D layout: lanes 0-15 -> M = r, lanes 16-31 -> M = 8 + r
                const int row = m0 + mt * 16 + half_id * 8 + r;
                float v = acc[mt][nt][r] + bias;
                if (!SCATTER) {
                    C[(size_t)row * ldc + n] = v;
                } else {                         // row = t*16 + b  ->  out[b][n][t]
                    const int bb = row & 15, tt = row >> 4;
                    C[((size_t)bb * V + n) * T + tt] = v;
                }
            }
        }
    }
}

// ---------------- persistent LSTM layer: full T-step recurrence on one WGP ----------------
// h kept in LDS as bf16 (A fragments via ds_load_b128); W_hh streams from L2.
// Phase 1 (per step): 16 waves tile Np gate columns in 32-col groups, NT=2 WMMA,
//   gates = h x W_hh^T + pregate_t  -> global scratch.
// Phase 2: fused gate nonlinearity, c update (global), h -> LDS bf16 + y row out.
__global__ __launch_bounds__(512) void lstm_layer_persistent(
    const float* __restrict__ pregate, int Np,      // (T*16, Np) f32
    const bf16_t* __restrict__ whh, int Kp,         // (Np, Kp) bf16
    float* __restrict__ gates,                      // (16, Np) f32 scratch
    float* __restrict__ c_state,                    // (16, Hh) f32 scratch (pre-zeroed)
    bf16_t* __restrict__ ybuf,                      // (T*16, Kp) bf16 layer output
    float* __restrict__ h_out, float* __restrict__ c_out,  // final states (16, Hh)
    int Hh) {
    __shared__ bf16_t hsh[16 * LDSW];               // 36 KB: recurrent h, padded
    const int tid     = threadIdx.x;
    const int lane    = tid & 31;
    const int wave    = tid >> 5;                   // 0..15
    const int lr      = lane & 15;
    const int half_id = lane >> 4;

    for (int i = tid; i < 16 * LDSW; i += 512) hsh[i] = (bf16_t)0.f;   // h0 = 0 (+ padding)
    __syncthreads();

    const int ngrp = Np / 32;                       // 32-col gate groups
    for (int t = 0; t < T; ++t) {
        const float* pre_t = pregate + (size_t)t * 16 * Np;

        // ---- phase 1: gates = h x W_hh^T + pre_t ----
        for (int grp = wave; grp < ngrp; grp += 16) {
            const int n0 = grp * 32;
            const bf16_t* arow  = hsh + (size_t)lr * LDSW;               // LDS
            const bf16_t* brow0 = whh + (size_t)(n0 + lr) * Kp;
            const bf16_t* brow1 = whh + (size_t)(n0 + 16 + lr) * Kp;
            v8f acc0 = {}, acc1 = {};
            for (int k0 = 0; k0 < Kp; k0 += 32) {
                v16bf a  = load_frag(arow, k0, half_id);                 // ds_load_b128
                v16bf b0 = load_frag(brow0, k0, half_id);
                v16bf b1 = load_frag(brow1, k0, half_id);
                acc0 = __builtin_amdgcn_wmma_f32_16x16x32_bf16(
                    false, a, false, b0, (short)0, acc0, false, false);
                acc1 = __builtin_amdgcn_wmma_f32_16x16x32_bf16(
                    false, a, false, b1, (short)0, acc1, false, false);
            }
#pragma unroll
            for (int r = 0; r < 8; ++r) {
                const int row = half_id * 8 + r;     // batch index b
                const size_t base = (size_t)row * Np;
                gates[base + n0 + lr]      = acc0[r] + pre_t[base + n0 + lr];
                gates[base + n0 + 16 + lr] = acc1[r] + pre_t[base + n0 + 16 + lr];
            }
        }
        __threadfence_block();
        __syncthreads();

        // ---- phase 2: nonlinearity + state update ----
        for (int idx = tid; idx < 16 * Hh; idx += 512) {
            const int b = idx / Hh, j = idx - b * Hh;
            const float* g = gates + (size_t)b * Np;
            const float ig = sigmoidf_fast(g[j]);
            const float fg = sigmoidf_fast(g[Hh + j]);
            const float gg = tanhf(g[2 * Hh + j]);
            const float og = sigmoidf_fast(g[3 * Hh + j]);
            const float c  = fg * c_state[idx] + ig * gg;
            const float h  = og * tanhf(c);
            c_state[idx] = c;
            hsh[b * LDSW + j] = (bf16_t)h;                               // next-step A
            ybuf[((size_t)t * 16 + b) * Kp + j] = (bf16_t)h;
            if (t == T - 1) { h_out[idx] = h; c_out[idx] = c; }
        }
        __threadfence_block();
        __syncthreads();
    }
}

// ---------------- f32 -> padded bf16 pack ----------------
__global__ void pack_bf16(const float* __restrict__ src, bf16_t* __restrict__ dst,
                          int rows_src, int cols_src, int ldd, long total) {
    const long idx = blockIdx.x * (long)blockDim.x + threadIdx.x;
    if (idx >= total) return;
    const long r = idx / ldd;
    const int  c = (int)(idx - r * ldd);
    float v = (r < rows_src && c < cols_src) ? src[r * (long)cols_src + c] : 0.f;
    dst[idx] = (bf16_t)v;
}

// ---------------- embedding gather -> padded bf16 (T,B,E) rows ----------------
__global__ void embed_gather(const float* __restrict__ emb, const int* __restrict__ x,
                             bf16_t* __restrict__ x0) {
    const int idx = blockIdx.x * blockDim.x + threadIdx.x;
    if (idx >= M * Ep) return;
    const int m = idx / Ep, e = idx - m * Ep;
    const int t = m >> 4, b = m & 15;
    const int tok = x[b * T + t];
    const float v = (e < E) ? emb[(size_t)tok * E + e] : 0.f;
    x0[idx] = (bf16_t)v;
}

// ---------------- host driver ----------------
extern "C" void kernel_launch(void* const* d_in, const int* in_sizes, int n_in,
                              void* d_out, int out_size, void* d_ws, size_t ws_size,
                              hipStream_t stream) {
    (void)in_sizes; (void)n_in; (void)out_size;

    const float* emb   = (const float*)d_in[0];
    const float* W_ih1 = (const float*)d_in[1];
    const float* W_hh1 = (const float*)d_in[2];
    const float* b_ih1 = (const float*)d_in[3];
    const float* b_hh1 = (const float*)d_in[4];
    const float* W_ih2 = (const float*)d_in[5];
    const float* W_hh2 = (const float*)d_in[6];
    const float* b_ih2 = (const float*)d_in[7];
    const float* b_hh2 = (const float*)d_in[8];
    const float* W_ih3 = (const float*)d_in[9];
    const float* W_hh3 = (const float*)d_in[10];
    const float* b_ih3 = (const float*)d_in[11];
    const float* b_hh3 = (const float*)d_in[12];
    const float* lin_b = (const float*)d_in[13];
    const int*   x     = (const int*)d_in[14];
    float* out = (float*)d_out;

    // workspace carve-up (256B aligned)
    size_t off = 0;
    auto take = [&](size_t bytes) -> char* {
        off = (off + 255) & ~(size_t)255;
        char* p = (char*)d_ws + off;
        off += bytes;
        return p;
    };
    bf16_t* emb_bf = (bf16_t*)take((size_t)V   * Ep * 2);
    bf16_t* wih1   = (bf16_t*)take((size_t)N1p * Ep * 2);
    bf16_t* whh1   = (bf16_t*)take((size_t)N1p * Hp * 2);
    bf16_t* wih2   = (bf16_t*)take((size_t)N1p * Hp * 2);
    bf16_t* whh2   = (bf16_t*)take((size_t)N1p * Hp * 2);
    bf16_t* wih3   = (bf16_t*)take((size_t)N3  * Hp * 2);
    bf16_t* whh3   = (bf16_t*)take((size_t)N3  * Ep * 2);
    bf16_t* x0     = (bf16_t*)take((size_t)M * Ep * 2);
    bf16_t* y1     = (bf16_t*)take((size_t)M * Hp * 2);
    bf16_t* y2     = (bf16_t*)take((size_t)M * Hp * 2);
    bf16_t* y3     = (bf16_t*)take((size_t)M * Ep * 2);
    float*  pregate= (float*) take((size_t)M * N1p * 4);
    float*  gates  = (float*) take((size_t)16 * N1p * 4);
    float*  c1s    = (float*) take((size_t)16 * H * 4);
    float*  c2s    = (float*) take((size_t)16 * H * 4);
    float*  c3s    = (float*) take((size_t)16 * E * 4);

    // zero all state/padding (bf16 zero == 0x0000)
    const size_t used = off;
    hipMemsetAsync(d_ws, 0, used <= ws_size ? used : ws_size, stream);

    // d_out state slots (return order: out, h1,h2,h3, c1,c2,c3)
    float* h1o = out + (size_t)B * V * T;
    float* h2o = h1o + (size_t)B * H;
    float* h3o = h2o + (size_t)B * H;
    float* c1o = h3o + (size_t)B * E;
    float* c2o = c1o + (size_t)B * H;
    float* c3o = c2o + (size_t)B * H;

    auto pack = [&](const float* s, bf16_t* d, int rs, int cs, int ldd, long rows_pad) {
        const long total = rows_pad * (long)ldd;
        pack_bf16<<<dim3((unsigned)((total + 255) / 256)), 256, 0, stream>>>(s, d, rs, cs, ldd, total);
    };
    pack(emb,   emb_bf, V,  E, Ep, V);
    pack(W_ih1, wih1,   N1, E, Ep, N1p);
    pack(W_hh1, whh1,   N1, H, Hp, N1p);
    pack(W_ih2, wih2,   N1, H, Hp, N1p);
    pack(W_hh2, whh2,   N1, H, Hp, N1p);
    pack(W_ih3, wih3,   N3, H, Hp, N3);
    pack(W_hh3, whh3,   N3, E, Ep, N3);

    embed_gather<<<dim3((M * Ep + 255) / 256), 256, 0, stream>>>(emb, x, x0);

    // one LSTM layer: time-batched pregate GEMM, then persistent on-chip recurrence
    auto run_layer = [&](const bf16_t* xin, int ldx,
                         const bf16_t* wih, const bf16_t* whh,
                         const float* bi, const float* bh,
                         int Hh, int Np, int Hh_p,
                         float* cbuf, bf16_t* ybuf,
                         float* h_out, float* c_out) {
        const int nbias = 4 * Hh;
        gemm_bf16_wmma<2, 4, 0><<<dim3((Np + 255) / 256, M / 32), 128, 0, stream>>>(
            xin, ldx, wih, ldx, bi, bh, nbias, pregate, Np, ldx, Np);
        lstm_layer_persistent<<<dim3(1), 512, 0, stream>>>(
            pregate, Np, whh, Hh_p, gates, cbuf, ybuf, h_out, c_out, Hh);
    };

    run_layer(x0, Ep, wih1, whh1, b_ih1, b_hh1, H, N1p, Hp, c1s, y1, h1o, c1o);
    run_layer(y1, Hp, wih2, whh2, b_ih2, b_hh2, H, N1p, Hp, c2s, y2, h2o, c2o);
    run_layer(y2, Hp, wih3, whh3, b_ih3, b_hh3, E, N3,  Ep, c3s, y3, h3o, c3o);

    // tied-weight logits: (2048 x Ep) * (V x Ep)^T -> scatter to (B, V, T)
    gemm_bf16_wmma<2, 4, 1><<<dim3(V / 256, M / 32), 128, 0, stream>>>(
        y3, Ep, emb_bf, Ep, lin_b, nullptr, V, out, 0, Ep, V);
}
